// SoftDecisionTree_84456236908591
// MI455X (gfx1250) — matrix operand order repaired and checked by reading, not compiled, not run on hardware
//
#include <hip/hip_runtime.h>

typedef _Float16 h16;
typedef __attribute__((ext_vector_type(16))) _Float16 v16h;
typedef __attribute__((ext_vector_type(8)))  float    v8f;
typedef __attribute__((ext_vector_type(4)))  float    v4f;

#define INPUT_DIM 256
#define NNODE     255
#define NLEAF     256
#define OUTD      32
#define WAVES     2          // waves per block (64 threads)
#define PSTRIDE   260        // padded prob row stride (floats); 8*260 % 64 != 0

#define NEG_LOG2E (-1.442695040888963f)

// ---- prep: Wh[n][k] = (-log2e * beta[n]) * W[n][k] as f16, 256x256, row 255 = 0 ----
__global__ void prep_w(const float* __restrict__ W, const float* __restrict__ beta,
                       h16* __restrict__ Wh) {
    int idx = blockIdx.x * 256 + threadIdx.x;     // 0..65535
    int n = idx >> 8;
    float v = (n < NNODE) ? (NEG_LOG2E * beta[n]) * W[idx] : 0.0f;
    Wh[idx] = (h16)v;
}

// ---- prep: QT[o][leaf] = softmax(leaf_param, axis=-1).T (f16); bb2 = -log2e*beta*b ----
__global__ void prep_misc(const float* __restrict__ leaf, const float* __restrict__ b,
                          const float* __restrict__ beta, h16* __restrict__ QT,
                          float* __restrict__ bb2) {
    int l = threadIdx.x;                          // 256 threads, one leaf each
    float vals[OUTD];
    float m = -1e30f;
#pragma unroll
    for (int o = 0; o < OUTD; ++o) { vals[o] = leaf[l * OUTD + o]; m = fmaxf(m, vals[o]); }
    float s = 0.0f;
#pragma unroll
    for (int o = 0; o < OUTD; ++o) {
        vals[o] = __builtin_amdgcn_exp2f((vals[o] - m) * 1.442695040888963f);
        s += vals[o];
    }
    float inv = 1.0f / s;
#pragma unroll
    for (int o = 0; o < OUTD; ++o) QT[o * NLEAF + l] = (h16)(vals[o] * inv);
    bb2[l] = (l < NNODE) ? NEG_LOG2E * beta[l] * b[l] : 0.0f;
}

// ---------------- main fused kernel: one 16-row batch tile per wave ----------------
__global__ __launch_bounds__(WAVES * 32, 1)
void tree_main(const float* __restrict__ x, const h16* __restrict__ Wh,
               const float* __restrict__ bb2, const h16* __restrict__ QT,
               float* __restrict__ out, float* __restrict__ pathp) {
    __shared__ float probs[WAVES][16 * PSTRIDE];
    __shared__ __align__(32) h16 qtile[OUTD * NLEAF];

    const int tid  = threadIdx.x;
    const int wave = tid >> 5;
    const int lane = tid & 31;
    const int h    = lane >> 4;       // half-wave id
    const int l16  = lane & 15;

    // stage QT (16 KB) into LDS, 16B per thread per iter
    {
        const uint4* src = (const uint4*)QT;
        uint4*       dst = (uint4*)qtile;
        for (int i = tid; i < (OUTD * NLEAF) / 8; i += WAVES * 32) dst[i] = src[i];
    }
    __syncthreads();

    const int row0 = (blockIdx.x * WAVES + wave) * 16;

    // ---- x rows -> f16 WMMA A fragments (contiguous k = 16h + s within each 32-k step); NT: x is streamed once ----
    const float* xrow = x + (size_t)(row0 + l16) * INPUT_DIM;
    v16h A[8];
#pragma unroll
    for (int kk = 0; kk < 8; ++kk) {
        const int kb = kk * 32 + h * 16;
        v4f c0 = __builtin_nontemporal_load((const v4f*)(xrow + kb));
        v4f c1 = __builtin_nontemporal_load((const v4f*)(xrow + kb + 4));
        v4f c2 = __builtin_nontemporal_load((const v4f*)(xrow + kb + 8));
        v4f c3 = __builtin_nontemporal_load((const v4f*)(xrow + kb + 12));
#pragma unroll
        for (int i = 0; i < 4; ++i) {
            A[kk][i]      = (h16)c0[i];
            A[kk][4 + i]  = (h16)c1[i];
            A[kk][8 + i]  = (h16)c2[i];
            A[kk][12 + i] = (h16)c3[i];
        }
    }

    float* Pw = probs[wave];

    // ---- GEMM1: scaled logits via WMMA, then prob = 1/(1+exp2(acc+bias)) -> LDS ----
#pragma unroll 2
    for (int nt = 0; nt < 16; ++nt) {
        const int n = nt * 16 + l16;                       // node column this lane owns
        const h16* wrow = Wh + n * INPUT_DIM + h * 16;     // contiguous 32B B-fragment rows
        v8f acc = {};
#pragma unroll
        for (int kk = 0; kk < 8; ++kk) {
            v16h Bf = *(const v16h*)(wrow + kk * 32);
            acc = __builtin_amdgcn_wmma_f32_16x16x32_f16(false, A[kk], false, Bf,
                                                         (short)0, acc, false, false);
        }
        const float bias = bb2[n];                         // already -log2e*beta*b
#pragma unroll
        for (int j = 0; j < 8; ++j) {                      // row = j + 8h
            float e  = __builtin_amdgcn_exp2f(acc[j] + bias);
            float pr = __builtin_amdgcn_rcpf(1.0f + e);    // sigmoid(beta*(logit+b))
            Pw[(j + 8 * h) * PSTRIDE + n] = pr;
        }
    }

    // ---- tree expansion: lane owns (row = l16, chunks c = 4(t>>1)+2h+(t&1) of 8 leaves) ----
    const float* P    = Pw + l16 * PSTRIDE;
    float*       prow = pathp + (size_t)(row0 + l16) * NLEAF;
    v16h pA[8];                                            // p as A-fragments for GEMM2
#pragma unroll
    for (int t = 0; t < 16; ++t) {
        const int c = 4 * (t >> 1) + 2 * h + (t & 1);      // leaf chunk: leaves c*8 .. c*8+7
        float base = 1.0f;                                 // shared top-5 path factors
#pragma unroll
        for (int d = 0; d < 5; ++d) {
            const int node = (1 << d) - 1 + (c >> (5 - d));
            const int bit  = (c >> (4 - d)) & 1;
            float f = P[node];
            base *= bit ? (1.0f - f) : f;
        }
        float p5  = P[31 + c];
        float p6a = P[63 + 2 * c],  p6b = P[64 + 2 * c];
        float q0  = P[127 + 4 * c], q1 = P[128 + 4 * c];
        float q2  = P[129 + 4 * c], q3 = P[130 + 4 * c];
        float t0 = base * p5,  t1 = base - t0;
        float u0 = t0 * p6a,   u1 = t0 - u0;
        float u2 = t1 * p6b,   u3 = t1 - u2;
        float L0 = u0 * q0, L1 = u0 - L0, L2 = u1 * q1, L3 = u1 - L2;
        float L4 = u2 * q2, L5 = u2 - L4, L6 = u3 * q3, L7 = u3 - L6;
        v4f s0 = {L0, L1, L2, L3}, s1 = {L4, L5, L6, L7};
        __builtin_nontemporal_store(s0, (v4f*)(prow + c * 8));      // path_probs (f32)
        __builtin_nontemporal_store(s1, (v4f*)(prow + c * 8 + 4));
        const int w = t >> 1, o = (t & 1) * 8;             // chunk == 4w+2h+s, s=t&1
        pA[w][o + 0] = (h16)L0; pA[w][o + 1] = (h16)L1;
        pA[w][o + 2] = (h16)L2; pA[w][o + 3] = (h16)L3;
        pA[w][o + 4] = (h16)L4; pA[w][o + 5] = (h16)L5;
        pA[w][o + 6] = (h16)L6; pA[w][o + 7] = (h16)L7;
    }

    // ---- GEMM2: out = p @ Q (Q columns contiguous in LDS qtile) ----
#pragma unroll
    for (int nt = 0; nt < 2; ++nt) {
        const h16* qrow = qtile + (nt * 16 + l16) * NLEAF + h * 16;
        v8f oacc = {};
#pragma unroll
        for (int kk = 0; kk < 8; ++kk) {
            v16h Bf = *(const v16h*)(qrow + kk * 32);
            oacc = __builtin_amdgcn_wmma_f32_16x16x32_f16(false, pA[kk], false, Bf,
                                                          (short)0, oacc, false, false);
        }
#pragma unroll
        for (int j = 0; j < 8; ++j)
            __builtin_nontemporal_store(
                oacc[j], out + (size_t)(row0 + j + 8 * h) * OUTD + nt * 16 + l16);
    }
}

extern "C" void kernel_launch(void* const* d_in, const int* in_sizes, int n_in,
                              void* d_out, int out_size, void* d_ws, size_t ws_size,
                              hipStream_t stream) {
    const float* x    = (const float*)d_in[0];   // (B, 256)
    const float* W    = (const float*)d_in[1];   // (255, 256)
    const float* b    = (const float*)d_in[2];   // (255,)
    const float* beta = (const float*)d_in[3];   // (255,)
    const float* leaf = (const float*)d_in[4];   // (256, 32)

    const int B = in_sizes[0] / INPUT_DIM;       // 131072

    float* out   = (float*)d_out;                      // (B, 32)
    float* pathp = (float*)d_out + (size_t)B * OUTD;   // (B, 256)

    // workspace carve-up (~148 KB)
    char*  ws  = (char*)d_ws;
    h16*   Wh  = (h16*)(ws);                     // 256*256*2 = 131072 B
    h16*   QT  = (h16*)(ws + 131072);            // 32*256*2  = 16384 B
    float* bb2 = (float*)(ws + 147456);          // 1024 B

    prep_w   <<<256, 256, 0, stream>>>(W, beta, Wh);
    prep_misc<<<1,   256, 0, stream>>>(leaf, b, beta, QT, bb2);

    const int rowsPerBlock = 16 * WAVES;
    tree_main<<<B / rowsPerBlock, WAVES * 32, 0, stream>>>(x, Wh, bb2, QT, out, pathp);
}
